// MultiEDSR_reshape_21500606283954
// MI455X (gfx1250) — compile-verified, matrix operand order checked by reading, not run on hardware
//
#include <hip/hip_runtime.h>

// ---------------------------------------------------------------------------
// Types for CDNA5 WMMA (gfx1250, wave32)
// ---------------------------------------------------------------------------
typedef __attribute__((ext_vector_type(16))) __bf16 v16bf;
typedef __attribute__((ext_vector_type(8)))  __bf16 v8bf;
typedef __attribute__((ext_vector_type(8)))  float  v8f;

static __device__ __forceinline__ unsigned short f2bf(float f) {
    // round-to-nearest-even fp32 -> bf16
    unsigned int u = __float_as_uint(f);
    unsigned int r = u + 0x7fffu + ((u >> 16) & 1u);
    return (unsigned short)(r >> 16);
}
static __device__ __forceinline__ __bf16 us_as_bf(unsigned short u) {
    union { unsigned short u; __bf16 h; } x; x.u = u; return x.h;
}

// Network constants
#define CINF   64                 // feature channels
#define KTOT   (CINF * 9)         // 576
#define KCHUNK ((CINF * 9) / 32)  // 18 chunks of K=32
#define NB     8                  // batch
#define HH     128
#define WW     128
#define CISTR  72                 // padded ci-stride in LDS (36 dwords -> bank rotate)

// K -> (lane-group, half) map for 16-bit WMMA fragments:
//   halves 0..7  <-> k = 8*hi + (0..7)
//   halves 8..15 <-> k = 16 + 8*hi + (0..7)
static __host__ __device__ __forceinline__ int kmap16(int lane, int h) {
    int v = h >> 1;
    int b = (v < 4) ? (2 * v) : (16 + 2 * (v - 4));
    return b + (((lane >> 4) & 1) << 3) + (h & 1);
}

// ---------------------------------------------------------------------------
// Weight repack: fp32 [COUT][CIN][3][3] -> bf16 in A-fragment-ready order.
// dst index = ((kc*MT + mt)*32 + lane)*16 + h ;  K order: k = tap*CIN + ci
// ---------------------------------------------------------------------------
__global__ void repack_weights_frag(const float* __restrict__ w,
                                    unsigned short* __restrict__ dst,
                                    int COUT, int COUTP) {
    int MT = COUTP >> 4;
    long long total = (long long)KCHUNK * MT * 512;
    long long idx = (long long)blockIdx.x * blockDim.x + threadIdx.x;
    if (idx >= total) return;
    int h    = (int)(idx & 15);
    int lane = (int)((idx >> 4) & 31);
    int mt   = (int)((idx >> 9) % MT);
    int kc   = (int)(idx / (512LL * MT));
    int k    = kc * 32 + kmap16(lane, h);
    int tap  = k / CINF;
    int ci   = k % CINF;
    int co   = mt * 16 + (lane & 15);
    float val = (co < COUT) ? w[(co * CINF + ci) * 9 + tap] : 0.0f;
    dst[idx] = f2bf(val);
}

// ---------------------------------------------------------------------------
// Implicit-GEMM 3x3 conv, CIN=64, via v_wmma_f32_16x16x32_bf16.
// Block: 256 threads (8 waves) -> 64 pixels (one row segment) x 64 couts.
// grid = (W/64, H, N * COUTP/64)
// LDS tile layout: s_in[(r*66 + x)*CISTR + ci] (bf16), contiguous in ci so
// each lane builds a B fragment with two ds_load_b128.
// ---------------------------------------------------------------------------
template <bool RELU, bool ADDRES, bool INBF, bool OUTBF, int COUT, int COUTP>
__global__ __launch_bounds__(256, 2)
void conv3x3_wmma(const void* __restrict__ in_,            // [N][64][H][W]
                  const unsigned short* __restrict__ wfrag, // fragment layout
                  const float* __restrict__ bias,           // [COUT]
                  const float* __restrict__ resid,          // [N][COUT][H][W]
                  void* __restrict__ out_) {                // [N][COUT][H][W]
    __shared__ __bf16 s_in[3 * 66 * CISTR];                 // 28,512 B

    constexpr int MT = COUTP >> 4;
    constexpr int coGroups = COUTP >> 6;
    const int w0 = blockIdx.x * 64;
    const int y  = blockIdx.y;
    const int n  = blockIdx.z / coGroups;
    const int cg = blockIdx.z % coGroups;

    // Stage 3 rows x 64 channels x 66 cols (zero-padded halo) as bf16.
    {
        const size_t inBase = (size_t)n * CINF * HH * WW;
        for (int i = threadIdx.x; i < 3 * CINF * 66; i += 256) {
            int c  = i % 66;          // x within padded tile
            int t  = i / 66;
            int ci = t % CINF;
            int r  = t / CINF;
            int gh = y + r - 1;
            int gw = w0 + c - 1;
            __bf16 v = us_as_bf(0);
            if ((unsigned)gh < (unsigned)HH && (unsigned)gw < (unsigned)WW) {
                size_t gi = inBase + ((size_t)ci * HH + gh) * WW + gw;
                if (INBF) v = us_as_bf(((const unsigned short*)in_)[gi]);
                else      v = us_as_bf(f2bf(((const float*)in_)[gi]));
            }
            s_in[(r * 66 + c) * CISTR + ci] = v;
        }
    }
    __syncthreads();

    const int lane   = threadIdx.x & 31;
    const int wave   = threadIdx.x >> 5;
    const int tileCo = wave & 3;        // 4 cout tiles of 16
    const int p0     = (wave >> 2) * 2; // 2 pixel tiles of 16 per wave
    const int mtg    = cg * 4 + tileCo; // global cout-tile index
    const int npix0  = p0 * 16 + (lane & 15);
    const int npix1  = npix0 + 16;
    const int hi8    = ((lane >> 4) & 1) << 3;
    const int xoff0  = npix0 * CISTR;   // LDS half-offset for pixel column
    const int xoff1  = npix1 * CISTR;

    v8f acc0 = {};
    v8f acc1 = {};

#pragma unroll 2
    for (int kc = 0; kc < KCHUNK; ++kc) {
        // A fragment (weights): one contiguous 32B global load per lane.
        const v16bf a = *(const v16bf*)(wfrag +
                          (((size_t)kc * MT + mtg) * 32 + lane) * 16);

        // B fragments: two ds_load_b128 per pixel tile.
        const int tap    = kc >> 1;        // CIN=64 -> one tap per 32-chunk
        const int ciBase = (kc & 1) * 32;
        const int kh = tap / 3, kw = tap % 3;
        const int off = (kh * 66 + kw) * CISTR + ciBase + hi8;

        v8bf lo0 = *(const v8bf*)&s_in[off + xoff0];
        v8bf hi0 = *(const v8bf*)&s_in[off + xoff0 + 16];
        v8bf lo1 = *(const v8bf*)&s_in[off + xoff1];
        v8bf hi1 = *(const v8bf*)&s_in[off + xoff1 + 16];
        v16bf b0 = __builtin_shufflevector(lo0, hi0,
                     0,1,2,3,4,5,6,7,8,9,10,11,12,13,14,15);
        v16bf b1 = __builtin_shufflevector(lo1, hi1,
                     0,1,2,3,4,5,6,7,8,9,10,11,12,13,14,15);

        acc0 = __builtin_amdgcn_wmma_f32_16x16x32_bf16(
                   false, a, false, b0, (short)0, acc0, false, false);
        acc1 = __builtin_amdgcn_wmma_f32_16x16x32_bf16(
                   false, a, false, b1, (short)0, acc1, false, false);
    }

    // Epilogue: D layout -> lane%16 = pixel (N), co = v + 8*(lane>=16) (M).
    const int coB = cg * 64 + tileCo * 16 + hi8;   // co for v = 0
    const int vmax = COUT - coB;                   // store v iff v < vmax
    const size_t HW = (size_t)HH * WW;
    const size_t base0 = (((size_t)n * COUT + coB) * HH + y) * WW + w0;
#pragma unroll
    for (int v = 0; v < 8; ++v) {
        if (v < vmax) {
            float bv = bias[coB + v];
            size_t o0 = base0 + (size_t)v * HW + npix0;
            size_t o1 = o0 + 16;
            float r0 = acc0[v] + bv;
            float r1 = acc1[v] + bv;
            if (ADDRES) { r0 += resid[o0]; r1 += resid[o1]; }
            if (RELU)   { r0 = fmaxf(r0, 0.0f); r1 = fmaxf(r1, 0.0f); }
            if (OUTBF) {
                ((unsigned short*)out_)[o0] = f2bf(r0);
                ((unsigned short*)out_)[o1] = f2bf(r1);
            } else {
                ((float*)out_)[o0] = r0;
                ((float*)out_)[o1] = r1;
            }
        }
    }
}

// ---------------------------------------------------------------------------
// Head conv: (x - mean) * head_w + head_b ; CIN=3, direct fp32.
// ---------------------------------------------------------------------------
__global__ void head_conv(const float* __restrict__ x,
                          const float* __restrict__ w,   // [64][3][3][3]
                          const float* __restrict__ b,
                          float* __restrict__ out) {
    const float kMean[3] = {255.0f * 0.4488f, 255.0f * 0.4371f, 255.0f * 0.404f};
    long long idx = (long long)blockIdx.x * blockDim.x + threadIdx.x;
    long long total = (long long)NB * 64 * HH * WW;
    if (idx >= total) return;
    int xw = (int)(idx % WW);
    int y  = (int)((idx / WW) % HH);
    int co = (int)((idx / ((long long)WW * HH)) % 64);
    int n  = (int)(idx / ((long long)WW * HH * 64));
    float acc = b[co];
    for (int ci = 0; ci < 3; ++ci) {
        for (int kh = 0; kh < 3; ++kh) {
            int gh = y + kh - 1;
            if ((unsigned)gh >= (unsigned)HH) continue;
            for (int kw = 0; kw < 3; ++kw) {
                int gw = xw + kw - 1;
                if ((unsigned)gw >= (unsigned)WW) continue;
                float v = x[(((size_t)n * 3 + ci) * HH + gh) * WW + gw] - kMean[ci];
                acc += v * w[((co * 3 + ci) * 3 + kh) * 3 + kw];
            }
        }
    }
    out[idx] = acc;
}

// ---------------------------------------------------------------------------
// Tail: pixel_shuffle(s=2) fused with 100->3 conv + add_mean.
// One thread = base pixel (y,x) -> produces the 2x2 output quad for all 3
// output channels; shares one 4x4 gathered patch per cc (16 loads vs 36).
// superOut: [N][400][128][128]; out: [N][3][256][256]
// ---------------------------------------------------------------------------
__global__ void tail_conv(const float* __restrict__ s,
                          const float* __restrict__ w,   // [3][100][3][3]
                          const float* __restrict__ b,
                          float* __restrict__ out) {
    const int OH = HH * 2, OW = WW * 2;
    const float kMean[3] = {255.0f * 0.4488f, 255.0f * 0.4371f, 255.0f * 0.404f};
    long long idx = (long long)blockIdx.x * blockDim.x + threadIdx.x;
    long long total = (long long)NB * HH * WW;
    if (idx >= total) return;
    int x = (int)(idx % WW);
    int y = (int)((idx / WW) % HH);
    int n = (int)(idx / ((long long)WW * HH));

    float acc[3][4];
#pragma unroll
    for (int c = 0; c < 3; ++c)
#pragma unroll
        for (int q = 0; q < 4; ++q) acc[c][q] = b[c];

    const size_t HW = (size_t)HH * WW;
    const float* sn = s + (size_t)n * 400 * HW;

    for (int cc = 0; cc < 100; ++cc) {
        const float* sc = sn + (size_t)cc * 4 * HW;
        // Gather 4x4 shuffled patch: rows ii = 2y-1..2y+2, cols jj = 2x-1..2x+2
        float g[4][4];
#pragma unroll
        for (int r = 0; r < 4; ++r) {
            int ii = 2 * y - 1 + r;
            bool rv = (unsigned)ii < (unsigned)OH;
            int si = ii >> 1;
            int subi = (ii & 1) * 2;
#pragma unroll
            for (int c2 = 0; c2 < 4; ++c2) {
                int jj = 2 * x - 1 + c2;
                float val = 0.0f;
                if (rv && (unsigned)jj < (unsigned)OW) {
                    int sub = subi + (jj & 1);
                    val = sc[(size_t)sub * HW + (size_t)si * WW + (jj >> 1)];
                }
                g[r][c2] = val;
            }
        }
#pragma unroll
        for (int c = 0; c < 3; ++c) {
            const float* wp = w + ((size_t)c * 100 + cc) * 9;  // wave-uniform
            float w9[9];
#pragma unroll
            for (int t = 0; t < 9; ++t) w9[t] = wp[t];
#pragma unroll
            for (int a = 0; a < 2; ++a)
#pragma unroll
                for (int b2 = 0; b2 < 2; ++b2)
#pragma unroll
                    for (int kh = 0; kh < 3; ++kh)
#pragma unroll
                        for (int kw = 0; kw < 3; ++kw)
                            acc[c][a * 2 + b2] += g[a + kh][b2 + kw] * w9[kh * 3 + kw];
        }
    }

    // Coalesced float2 stores: row 2y+a, cols (2x, 2x+1).
#pragma unroll
    for (int c = 0; c < 3; ++c) {
#pragma unroll
        for (int a = 0; a < 2; ++a) {
            float2 v2 = make_float2(acc[c][a * 2 + 0] + kMean[c],
                                    acc[c][a * 2 + 1] + kMean[c]);
            size_t o = (((size_t)n * 3 + c) * OH + (2 * y + a)) * OW + 2 * x;
            *(float2*)(out + o) = v2;
        }
    }
}

// ---------------------------------------------------------------------------
// Launch
// ---------------------------------------------------------------------------
extern "C" void kernel_launch(void* const* d_in, const int* in_sizes, int n_in,
                              void* d_out, int out_size, void* d_ws, size_t ws_size,
                              hipStream_t stream) {
    const float* x       = (const float*)d_in[0];
    const float* head_w  = (const float*)d_in[1];
    const float* head_b  = (const float*)d_in[2];
    const float* rb_w1   = (const float*)d_in[3];
    const float* rb_b1   = (const float*)d_in[4];
    const float* rb_w2   = (const float*)d_in[5];
    const float* rb_b2   = (const float*)d_in[6];
    const float* body_w  = (const float*)d_in[7];
    const float* body_b  = (const float*)d_in[8];
    const float* super_w = (const float*)d_in[9];
    const float* super_b = (const float*)d_in[10];
    const float* out_w   = (const float*)d_in[11];
    const float* out_b   = (const float*)d_in[12];
    float* y = (float*)d_out;

    // Workspace layout
    const size_t WB_SLOT = (size_t)KTOT * 64;            // bf16 elems per 64x64 conv
    const size_t SUPER_P = 448;                          // 400 padded to 448
    const size_t WB_SUPER = (size_t)KTOT * SUPER_P;
    unsigned short* wb       = (unsigned short*)d_ws;    // 33 slots
    unsigned short* wb_super = wb + 33 * WB_SLOT;
    size_t wbytes = (33 * WB_SLOT + WB_SUPER) * sizeof(unsigned short);
    wbytes = (wbytes + 255) & ~(size_t)255;
    float* fbase = (float*)((char*)d_ws + wbytes);
    const size_t ACT = (size_t)NB * 64 * HH * WW;        // 8.39M elems
    float* bufH = fbase;                                 // head out (skip)
    float* bufA = bufH + ACT;                            // carry ping
    unsigned short* bufT = (unsigned short*)(bufA + ACT);// resblock tmp (bf16)
    float* bufC = (float*)(bufT + ACT);                  // carry pong
    float* bufB = bufC + ACT;                            // body out (fp32)
    float* bufS = bufB + ACT;                            // [N][400][128][128]

    // 1) Repack all conv weights into WMMA fragment order (bf16).
    {
        int total64 = KCHUNK * 4 * 512;                  // 36,864
        dim3 g64((total64 + 255) / 256), blk(256);
        for (int s = 0; s < 16; ++s) {
            repack_weights_frag<<<g64, blk, 0, stream>>>(
                rb_w1 + (size_t)s * 64 * 64 * 9, wb + (size_t)s * WB_SLOT, 64, 64);
            repack_weights_frag<<<g64, blk, 0, stream>>>(
                rb_w2 + (size_t)s * 64 * 64 * 9, wb + (size_t)(16 + s) * WB_SLOT, 64, 64);
        }
        repack_weights_frag<<<g64, blk, 0, stream>>>(body_w, wb + 32 * WB_SLOT, 64, 64);
        long long totalS = (long long)KCHUNK * (SUPER_P / 16) * 512; // 258,048
        repack_weights_frag<<<dim3((unsigned)((totalS + 255) / 256)), blk, 0, stream>>>(
            super_w, wb_super, 400, (int)SUPER_P);
    }

    // 2) Head conv -> bufH (fp32)
    {
        long long total = (long long)NB * 64 * HH * WW;
        head_conv<<<dim3((unsigned)((total + 255) / 256)), dim3(256), 0, stream>>>(
            x, head_w, head_b, bufH);
    }

    // 3) 16 resblocks (WMMA). conv1: fp32 in -> bf16 tmp (identical rounding
    //    to what conv2's bf16 staging would do). conv2: bf16 in -> fp32 +
    //    residual. Carry ping-pongs bufA/bufC; bufH preserved for body skip.
    dim3 cgrid(WW / 64, HH, NB);
    dim3 cblk(256);
    const float* cur = bufH;
    float* pp[2] = {bufA, bufC};
    for (int i = 0; i < 16; ++i) {
        float* nxt = pp[i & 1];
        conv3x3_wmma<true,  false, false, true, 64, 64><<<cgrid, cblk, 0, stream>>>(
            cur, wb + (size_t)i * WB_SLOT, rb_b1 + (size_t)i * 64, nullptr, bufT);
        conv3x3_wmma<false, true,  true, false, 64, 64><<<cgrid, cblk, 0, stream>>>(
            bufT, wb + (size_t)(16 + i) * WB_SLOT, rb_b2 + (size_t)i * 64, cur, nxt);
        cur = nxt;
    }

    // 4) Body conv + skip(h) -> bufB (fp32)
    conv3x3_wmma<false, true, false, false, 64, 64><<<cgrid, cblk, 0, stream>>>(
        cur, wb + 32 * WB_SLOT, body_b, bufH, bufB);

    // 5) Super conv 64 -> 400 (padded 448) -> bufS (fp32)
    dim3 sgrid(WW / 64, HH, NB * (int)(SUPER_P / 64));
    conv3x3_wmma<false, false, false, false, 400, 448><<<sgrid, cblk, 0, stream>>>(
        bufB, wb_super, super_b, nullptr, bufS);

    // 6) Fused pixel-shuffle + 100->3 conv + add_mean -> d_out
    {
        long long total = (long long)NB * HH * WW;
        tail_conv<<<dim3((unsigned)((total + 255) / 256)), dim3(256), 0, stream>>>(
            bufS, out_w, out_b, y);
    }
}
